// Attention_12567074308600
// MI455X (gfx1250) — matrix operand order repaired and checked by reading, not compiled, not run on hardware
//
#include <hip/hip_runtime.h>

#define DIM      384
#define INNER3   1152
#define HEADS    12
#define DH       32
#define S_TOK    64
#define SCALE    0.17677669529663689f   // 32^-0.5

// LDS strides (in halves), padded to break bank conflicts, rows 16B aligned
#define XST  392   // x / ctx buffer
#define QST  392
#define KTST 72    // K^T: [head*32 + d][token]
#define VST  392
#define AST  72    // per-wave attn staging 16x64

#define OFF_Q   25088
#define OFF_KT  50176
#define OFF_V   77824
#define OFF_A   102912
#define LDS_H   112128   // total halves = 224,256 bytes

typedef _Float16 h8  __attribute__((ext_vector_type(8)));
typedef _Float16 v16h __attribute__((ext_vector_type(16)));
typedef float    v8f  __attribute__((ext_vector_type(8)));

static __device__ __forceinline__ v8f wmma_f16(v16h a, v16h b, v8f c) {
  return __builtin_amdgcn_wmma_f32_16x16x32_f16(false, a, false, b, (short)0, c,
                                                false, false);
}

// A-matrix tile (16x32 f16): lane holds row m = row0 + (lane&15);
// halves 0..7 = K {0..7 | 8..15}, halves 8..15 = +16 (ISA 7.12.2 layout)
static __device__ __forceinline__ v16h load_tileA(const _Float16* base, int stride,
                                                  int row0, int col0, int lane) {
  const int m = row0 + (lane & 15);
  const int k = col0 + ((lane >> 4) << 3);
  const _Float16* p = base + m * stride + k;
  h8 lo = *(const h8*)p;
  h8 hi = *(const h8*)(p + 16);
  return __builtin_shufflevector(lo, hi, 0,1,2,3,4,5,6,7,8,9,10,11,12,13,14,15);
}

// B-matrix tile (32x16 f16): lane holds column n = n0 + (lane&15);
// halves 0..7 = K {0..7 | 8..15}, halves 8..15 = +16 (same as A)
static __device__ __forceinline__ v16h load_tileB(const _Float16* base, int stride,
                                                  int k0, int n0, int lane) {
  const _Float16* p = base + (k0 + ((lane >> 4) << 3)) * stride + n0 + (lane & 15);
  v16h b;
#pragma unroll
  for (int s = 0; s < 8; ++s) {
    b[s]     = p[s * stride];
    b[s + 8] = p[(s + 16) * stride];
  }
  return b;
}

// One-time weight downconvert into workspace (weights then live in 192MB L2)
__global__ void cvt_weights_kernel(const float* __restrict__ wqkv,
                                   const float* __restrict__ wout,
                                   _Float16* __restrict__ dst) {
  int idx = blockIdx.x * 256 + threadIdx.x;
  if (idx < DIM * INNER3) {
    dst[idx] = (_Float16)wqkv[idx];
  } else if (idx < DIM * INNER3 + DIM * DIM) {
    dst[idx] = (_Float16)(wout[idx - DIM * INNER3]);
  }
}

__global__ __launch_bounds__(256, 1)
void attn_win_kernel(const float* __restrict__ x,
                     const _Float16* __restrict__ wqkv_h,
                     const _Float16* __restrict__ wout_h,
                     const float* __restrict__ b_out,
                     float* __restrict__ out) {
  __shared__ __align__(16) _Float16 smem[LDS_H];
  _Float16* Xs  = smem;             // 64 x 384 (stride 392); reused as ctx
  _Float16* Qs  = smem + OFF_Q;     // 64 x 384, pre-scaled
  _Float16* Kts = smem + OFF_KT;    // [h*32+d][token], stride 72
  _Float16* Vs  = smem + OFF_V;     // 64 x 384
  _Float16* As  = smem + OFF_A;     // 8 waves x (16 x 64, stride 72)

  const int tid  = threadIdx.x;
  const int wave = tid >> 5;
  const int lane = tid & 31;
  const int ln15 = lane & 15;
  const int mb   = (lane >> 4) << 3;   // 0 or 8: C-layout row offset

  // window id -> base pixel index
  const int n  = blockIdx.x;           // 0..2047
  const int bb = n >> 6;
  const int wh = (n >> 3) & 7;
  const int ww = n & 7;
  const int rowBase = (bb * 64 + wh * 8) * 64 + ww * 8; // pixel of token (0,0)

  // ---------------- stage 0: gather window, f32 -> f16 LDS ----------------
  for (int idx = tid; idx < S_TOK * 96; idx += 256) {
    const int t  = idx / 96;
    const int c4 = (idx - t * 96) * 4;
    const int pix = rowBase + ((t >> 3) << 6) + (t & 7);
    const float4 v = *(const float4*)(x + pix * DIM + c4);
    _Float16* d = Xs + t * XST + c4;
    d[0] = (_Float16)v.x; d[1] = (_Float16)v.y;
    d[2] = (_Float16)v.z; d[3] = (_Float16)v.w;
  }
  __syncthreads();

  // -------- stage 1: QKV GEMM [64x384] @ [384x1152], 3 N-tiles/group ------
  // Wave handles nt = wave + 8*i, i = 0..8; group g covers i = 3g..3g+2,
  // i.e. nt = wave + 24g + 8j, which lies entirely in section g (Q/K/V).
  for (int g = 0; g < 3; ++g) {
    const int nt0 = wave + g * 24;
    v8f acc[3][4] = {};
#pragma unroll 4
    for (int k = 0; k < 12; ++k) {
      v16h bt[3];
#pragma unroll
      for (int j = 0; j < 3; ++j)
        bt[j] = load_tileB(wqkv_h, INNER3, k * 32, (nt0 + 8 * j) * 16, lane);
#pragma unroll
      for (int mt = 0; mt < 4; ++mt) {
        const v16h at = load_tileA(Xs, XST, mt * 16, k * 32, lane);
#pragma unroll
        for (int j = 0; j < 3; ++j)
          acc[j][mt] = wmma_f16(at, bt[j], acc[j][mt]);
      }
    }
    const int sec = g;                  // section is uniform per group
#pragma unroll
    for (int j = 0; j < 3; ++j) {
      const int c  = (nt0 + 8 * j) * 16 + ln15;   // 0..1151
      const int cc = c - sec * 384;               // 0..383 = h*32 + d
#pragma unroll
      for (int mt = 0; mt < 4; ++mt) {
#pragma unroll
        for (int r = 0; r < 8; ++r) {
          const int m = mt * 16 + r + mb;
          const float v = acc[j][mt][r];
          if (sec == 0)      Qs[m * QST + cc]   = (_Float16)(v * SCALE);
          else if (sec == 1) Kts[cc * KTST + m] = (_Float16)v;   // transposed
          else               Vs[m * VST + cc]   = (_Float16)v;
        }
      }
    }
  }
  __syncthreads();

  // ------------- stage 2: attention, 48 (head, mtile) units / 8 waves ------
  _Float16* Aw = As + wave * (16 * AST);
  for (int ui = 0; ui < 6; ++ui) {
    const int u  = wave + ui * 8;       // 0..47
    const int h  = u >> 2;
    const int mt = u & 3;
    const int hc = h * 32;

    // sim = (Q*scale) @ K^T : 4 WMMAs (K = dim_head = 32)
    const v16h aq = load_tileA(Qs, QST, mt * 16, hc, lane);
    v8f sim[4];
#pragma unroll
    for (int nt2 = 0; nt2 < 4; ++nt2) {
      const v16h bk = load_tileB(Kts + hc * KTST, KTST, 0, nt2 * 16, lane);
      v8f z = {};
      sim[nt2] = wmma_f16(aq, bk, z);
    }

    // row softmax: row lives in one 16-lane half (4 vals in-lane over nt2)
#pragma unroll
    for (int r = 0; r < 8; ++r) {
      float mx = fmaxf(fmaxf(sim[0][r], sim[1][r]), fmaxf(sim[2][r], sim[3][r]));
#pragma unroll
      for (int off = 8; off >= 1; off >>= 1)
        mx = fmaxf(mx, __shfl_xor(mx, off, 32));
      const float e0 = __expf(sim[0][r] - mx);
      const float e1 = __expf(sim[1][r] - mx);
      const float e2 = __expf(sim[2][r] - mx);
      const float e3 = __expf(sim[3][r] - mx);
      float s = e0 + e1 + e2 + e3;
#pragma unroll
      for (int off = 8; off >= 1; off >>= 1)
        s += __shfl_xor(s, off, 32);
      const float inv = 1.0f / s;
      _Float16* d = Aw + (r + mb) * AST + ln15;  // redistribute C->A layout
      d[0]  = (_Float16)(e0 * inv);
      d[16] = (_Float16)(e1 * inv);
      d[32] = (_Float16)(e2 * inv);
      d[48] = (_Float16)(e3 * inv);
    }
    asm volatile("s_wait_dscnt 0" ::: "memory");

    // out_head = attn @ V : K = 64 tokens -> 2 accum steps, N = 32 -> 2 tiles
    v8f o[2] = {};
#pragma unroll
    for (int kt2 = 0; kt2 < 2; ++kt2) {
      const v16h aa = load_tileA(Aw, AST, 0, kt2 * 32, lane);
#pragma unroll
      for (int nd = 0; nd < 2; ++nd) {
        const v16h bv = load_tileB(Vs, VST, kt2 * 32, hc + nd * 16, lane);
        o[nd] = wmma_f16(aa, bv, o[nd]);
      }
    }
#pragma unroll
    for (int nd = 0; nd < 2; ++nd) {
      const int c2 = hc + nd * 16 + ln15;
#pragma unroll
      for (int r = 0; r < 8; ++r) {
        const int m = mt * 16 + r + mb;
        Xs[m * XST + c2] = (_Float16)o[nd][r];  // ctx reuses x region
      }
    }
  }
  __syncthreads();

  // ------- stage 3: out GEMM [64x384] @ [384x384] + bias, 3 N-tiles -------
  {
    v8f acc[3][4] = {};
#pragma unroll 4
    for (int k = 0; k < 12; ++k) {
      v16h bt[3];
#pragma unroll
      for (int j = 0; j < 3; ++j)
        bt[j] = load_tileB(wout_h, DIM, k * 32, (wave + 8 * j) * 16, lane);
#pragma unroll
      for (int mt = 0; mt < 4; ++mt) {
        const v16h at = load_tileA(Xs, XST, mt * 16, k * 32, lane);
#pragma unroll
        for (int j = 0; j < 3; ++j)
          acc[j][mt] = wmma_f16(at, bt[j], acc[j][mt]);
      }
    }
#pragma unroll
    for (int j = 0; j < 3; ++j) {
      const int c = (wave + 8 * j) * 16 + ln15;
      const float bias = b_out[c];
#pragma unroll
      for (int mt = 0; mt < 4; ++mt) {
#pragma unroll
        for (int r = 0; r < 8; ++r) {
          const int m   = mt * 16 + r + mb;
          const int pix = rowBase + ((m >> 3) << 6) + (m & 7);
          out[pix * DIM + c] = acc[j][mt][r] + bias;
        }
      }
    }
  }
}

extern "C" void kernel_launch(void* const* d_in, const int* in_sizes, int n_in,
                              void* d_out, int out_size, void* d_ws, size_t ws_size,
                              hipStream_t stream) {
  (void)in_sizes; (void)n_in; (void)out_size; (void)ws_size;
  const float* x    = (const float*)d_in[0];
  const float* wqkv = (const float*)d_in[1];
  const float* wout = (const float*)d_in[2];
  const float* bout = (const float*)d_in[3];

  _Float16* wbuf = (_Float16*)d_ws;   // [0,442368) = w_qkv f16, then w_out f16
  const int total = DIM * INNER3 + DIM * DIM;          // 589,824
  cvt_weights_kernel<<<(total + 255) / 256, 256, 0, stream>>>(wqkv, wout, wbuf);
  attn_win_kernel<<<2048, 256, 0, stream>>>(x, wbuf, wbuf + DIM * INNER3, bout,
                                            (float*)d_out);
}
